// COGConv2d_71476845740326
// MI455X (gfx1250) — matrix operand order repaired and checked by reading, not compiled, not run on hardware
//
#include <hip/hip_runtime.h>
#include <math.h>

typedef __bf16 bf16_t;
typedef __attribute__((ext_vector_type(16))) __bf16 v16bf;
typedef __attribute__((ext_vector_type(8)))  __bf16 v8bf;
typedef __attribute__((ext_vector_type(8)))  float  v8f;
typedef __attribute__((ext_vector_type(4)))  int    v4i_t;

#define BATCH 32
#define CIN   256
#define COUT  256
#define KS    3
#define HH    56
#define WW    56
#define CHK   4
#define HP    58       // padded spatial (pad=1)
#define NPIX  3136     // 56*56

// ---- gfx1250 async global->LDS path (hedged: fall back if builtin absent) ----
#if defined(__has_builtin)
#  if __has_builtin(__builtin_amdgcn_global_load_async_to_lds_b128)
#    define HAVE_ASYNC_LDS 1
#  endif
#  if __has_builtin(__builtin_amdgcn_s_wait_asynccnt)
#    define HAVE_WAIT_ASYNC 1
#  endif
#endif
#ifndef HAVE_ASYNC_LDS
#  define HAVE_ASYNC_LDS 0
#endif
#ifndef HAVE_WAIT_ASYNC
#  define HAVE_WAIT_ASYNC 0
#endif

// ---------------- kernel 0: global average pool -> pooled[B*C] ----------------
__global__ void pool_kernel(const float* __restrict__ x, float* __restrict__ pooled) {
    __shared__ float red[256];
    const int bc = blockIdx.x;                 // b*C + c
    const float* p = x + (size_t)bc * NPIX;
    float s = 0.f;
    for (int i = threadIdx.x; i < NPIX; i += 256) s += p[i];
    red[threadIdx.x] = s;
    __syncthreads();
    for (int off = 128; off > 0; off >>= 1) {
        if (threadIdx.x < off) red[threadIdx.x] += red[threadIdx.x + off];
        __syncthreads();
    }
    if (threadIdx.x == 0) pooled[bc] = red[0] * (1.0f / (float)NPIX);
}

// ---------------- kernel 1: h = relu(pooled @ fc1_w^T) ----------------
__global__ void fc1_kernel(const float* __restrict__ pooled,
                           const float* __restrict__ fc1_w,
                           float* __restrict__ h) {
    const int b = blockIdx.x;
    const int o = threadIdx.x;                 // 256 threads
    const float* pv = pooled + b * CIN;
    const float* wv = fc1_w + o * CIN;
    float s = 0.f;
    for (int c = 0; c < CIN; ++c) s += pv[c] * wv[c];
    h[b * CIN + o] = s > 0.f ? s : 0.f;
}

// ---------------- kernel 2: kern = h @ fc2_w^T + fc2_b ----------------
__global__ void fc2_kernel(const float* __restrict__ h,
                           const float* __restrict__ fc2_w,
                           const float* __restrict__ fc2_b,
                           float* __restrict__ kern) {
    const int idx = blockIdx.x * 256 + threadIdx.x;   // B*1024 = 32768
    const int b  = idx >> 10;
    const int oc = idx & 1023;
    const float* hv = h + b * CIN;
    const float* wv = fc2_w + oc * CIN;
    float s = fc2_b[oc];
    for (int c = 0; c < CIN; ++c) s += hv[c] * wv[c];
    kern[idx] = s;
}

// ------- kernel 3: dynamic weights, bf16, tap-major A[B][9][O][C] -------
__global__ void dynw_kernel(const float* __restrict__ kern,
                            const float* __restrict__ cogw,   // [O][CH][3][3]
                            const float* __restrict__ w,      // [O][C][3][3]
                            bf16_t* __restrict__ A) {
    const int idx = blockIdx.x * 256 + threadIdx.x;  // B*9*O*C = 18,874,368
    const int c    = idx & 255;
    const int o    = (idx >> 8) & 255;
    const int rest = idx >> 16;                       // b*9 + tap
    const int tap  = rest % 9;
    const int b    = rest / 9;
    const float* kv = kern + (b << 10) + (c << 2);
    const float* cg = cogw + o * (CHK * 9) + tap;
    float cw = 0.f;
#pragma unroll
    for (int t = 0; t < CHK; ++t) cw += kv[t] * cg[t * 9];
    const float sg = 1.f / (1.f + __expf(-cw));
    const float dw = sg * w[(size_t)(o * CIN + c) * 9 + tap];
    A[idx] = (bf16_t)dw;                              // [b][tap][o][c]
}

// ---------------- kernel 4: padded bf16 copy of x ----------------
__global__ void xpad_kernel(const float* __restrict__ x, bf16_t* __restrict__ xp) {
    const int idx = blockIdx.x * 256 + threadIdx.x;   // B*C*58*58 = 27,557,888
    const int xc = idx % HP;
    const int t  = idx / HP;
    const int y  = t % HP;
    const int bc = t / HP;                            // b*C + c
    float v = 0.f;
    if (y >= 1 && y <= HH && xc >= 1 && xc <= WW)
        v = x[(size_t)bc * NPIX + (y - 1) * WW + (xc - 1)];
    xp[idx] = (bf16_t)v;
}

// ---------------- kernel 5: implicit-GEMM conv via bf16 WMMA ----------------
// Per sample b: out[O=256, N=3136] = sum_{tap} A_tap[O,C] * B_tap[C,N]
// Block: 256 threads = 8 waves. Block tile M=128, N=64, K-step 32.
// Wave w: m_block = w>>1 (4 blocks of 32), n_block = w&1 (2 blocks of 32)
//         -> 2x2 WMMA tiles (16x16) per wave per K-step.
__global__ __launch_bounds__(256) void cog_conv_wmma(
    const bf16_t* __restrict__ A,   // [B][9][O][C]
    const bf16_t* __restrict__ xp,  // [B][C][58][58]
    float* __restrict__ out)        // [B][O][56][56]
{
    __shared__ bf16_t As[128][32];  // [m_local][k_local]  8 KB
    __shared__ bf16_t Bs[64][32];   // [n_local][k_local]  4 KB (transposed staging)

    const int b   = blockIdx.z;
    const int mo  = blockIdx.y * 128;      // O tile base   (2 tiles)
    const int n0  = blockIdx.x * 64;       // N tile base   (49 tiles)
    const int tid = threadIdx.x;
    const int lane    = tid & 31;
    const int wave    = tid >> 5;
    const int half    = lane >> 4;         // K-half select (wave32 layout)
    const int l16     = lane & 15;
    const int m_block = wave >> 1;         // 0..3  -> M offset 32*m_block
    const int n_block = wave & 1;          // 0..1  -> N offset 32*n_block

    v8f acc00 = {}, acc01 = {}, acc10 = {}, acc11 = {};

    const bf16_t* Abase = A  + (size_t)b * 9 * COUT * CIN;
    const bf16_t* Xbase = xp + (size_t)b * CIN * HP * HP;

    for (int tap = 0; tap < 9; ++tap) {
        const int di = tap / 3;
        const int dj = tap - 3 * di;
        const bf16_t* Atap = Abase + (size_t)tap * COUT * CIN;

        for (int kc = 0; kc < CIN; kc += 32) {
            // ---- stage A tile 128x32: 512 chunks of 8 bf16 (16 B), 2/thread ----
#pragma unroll
            for (int r = 0; r < 2; ++r) {
                const int ch  = r * 256 + tid;
                const int row = ch >> 2;
                const int col = (ch & 3) * 8;
                const bf16_t* gp = Atap + (size_t)(mo + row) * CIN + (kc + col);
#if HAVE_ASYNC_LDS
                __builtin_amdgcn_global_load_async_to_lds_b128(
                    (__attribute__((address_space(1))) v4i_t*)gp,
                    (__attribute__((address_space(3))) v4i_t*)&As[row][col],
                    0, 0);
#else
                *(v8bf*)&As[row][col] = *(const v8bf*)gp;
#endif
            }
            // prefetch next A tile into L2 (global_prefetch_b8)
            if (kc + 32 < CIN)
                __builtin_prefetch(Atap + (size_t)(mo + (tid >> 1)) * CIN + kc + 32, 0, 1);

            // ---- stage B tile: gather shifted x windows into Bs[n][k] ----
#pragma unroll
            for (int r = 0; r < 8; ++r) {
                const int idx  = r * 256 + tid;
                const int nloc = idx >> 5, kloc = idx & 31;
                const int n  = n0 + nloc;
                const int yy = n / WW;
                const int xx = n - yy * WW;
                Bs[nloc][kloc] =
                    Xbase[((size_t)(kc + kloc) * HP + (yy + di)) * HP + (xx + dj)];
            }
#if HAVE_ASYNC_LDS && HAVE_WAIT_ASYNC
            __builtin_amdgcn_s_wait_asynccnt(0);
#endif
            __syncthreads();

            // ---- build WMMA fragments per documented 16-bit layouts ----
            v16bf a0, a1, bf0, bf1;
            const int mr0 = m_block * 32 + l16;
            const int mr1 = mr0 + 16;
            const int nr0 = n_block * 32 + l16;
            const int nr1 = nr0 + 16;
#pragma unroll
            for (int e = 0; e < 16; ++e) {
                const int kA = (e < 8) ? (8 * half + e) : (8 * half + 8 + e);
                a0[e]  = As[mr0][kA];
                a1[e]  = As[mr1][kA];
                const int kB = (e < 8) ? (8 * half + e) : (8 * half + 8 + e);
                bf0[e] = Bs[nr0][kB];
                bf1[e] = Bs[nr1][kB];
            }

            acc00 = __builtin_amdgcn_wmma_f32_16x16x32_bf16(
                false, a0, false, bf0, (short)0, acc00, false, false);
            acc01 = __builtin_amdgcn_wmma_f32_16x16x32_bf16(
                false, a0, false, bf1, (short)0, acc01, false, false);
            acc10 = __builtin_amdgcn_wmma_f32_16x16x32_bf16(
                false, a1, false, bf0, (short)0, acc10, false, false);
            acc11 = __builtin_amdgcn_wmma_f32_16x16x32_bf16(
                false, a1, false, bf1, (short)0, acc11, false, false);

            __syncthreads();
        }
    }

    // ---- epilogue: 16x16 f32 C layout (vgpr r -> M=r+8*half, N=lane&15) ----
    const int mg0 = mo + m_block * 32 + 8 * half;
    const int mg1 = mg0 + 16;
    const int ng0 = n0 + n_block * 32 + l16;
    const int ng1 = ng0 + 16;
    float* ob = out + (size_t)b * COUT * NPIX;
#pragma unroll
    for (int r = 0; r < 8; ++r) {
        ob[(size_t)(mg0 + r) * NPIX + ng0] = acc00[r];
        ob[(size_t)(mg0 + r) * NPIX + ng1] = acc01[r];
        ob[(size_t)(mg1 + r) * NPIX + ng0] = acc10[r];
        ob[(size_t)(mg1 + r) * NPIX + ng1] = acc11[r];
    }
}

extern "C" void kernel_launch(void* const* d_in, const int* in_sizes, int n_in,
                              void* d_out, int out_size, void* d_ws, size_t ws_size,
                              hipStream_t stream) {
    const float* x     = (const float*)d_in[0];
    const float* fc1_w = (const float*)d_in[1];
    const float* fc2_w = (const float*)d_in[2];
    const float* fc2_b = (const float*)d_in[3];
    const float* cogw  = (const float*)d_in[4];
    const float* w     = (const float*)d_in[5];
    float* out = (float*)d_out;

    // workspace layout (all 256B aligned)
    char* ws = (char*)d_ws;
    bf16_t* xp = (bf16_t*)ws;
    ws += (size_t)BATCH * CIN * HP * HP * sizeof(bf16_t);       // 55,115,776
    bf16_t* Abuf = (bf16_t*)ws;
    ws += (size_t)BATCH * 9 * COUT * CIN * sizeof(bf16_t);      // 37,748,736
    float* pooled = (float*)ws; ws += (size_t)BATCH * CIN * sizeof(float);
    float* h      = (float*)ws; ws += (size_t)BATCH * CIN * sizeof(float);
    float* kern   = (float*)ws; ws += (size_t)BATCH * CHK * CIN * sizeof(float);

    // 0) pool
    pool_kernel<<<BATCH * CIN, 256, 0, stream>>>(x, pooled);
    // 1) fc1 + relu
    fc1_kernel<<<BATCH, 256, 0, stream>>>(pooled, fc1_w, h);
    // 2) fc2 + bias
    fc2_kernel<<<(BATCH * CHK * CIN) / 256, 256, 0, stream>>>(h, fc2_w, fc2_b, kern);
    // 3) dynamic weights -> bf16 tap-major A
    dynw_kernel<<<(BATCH * 9 * COUT * CIN) / 256, 256, 0, stream>>>(kern, cogw, w, Abuf);
    // 4) padded bf16 x
    xpad_kernel<<<(BATCH * CIN * HP * HP) / 256, 256, 0, stream>>>(x, xp);
    // 5) main WMMA implicit-GEMM conv
    dim3 grid(NPIX / 64, COUT / 128, BATCH);
    cog_conv_wmma<<<grid, 256, 0, stream>>>(Abuf, xp, out);
}
